// RQKmeans_54992761258888
// MI455X (gfx1250) — matrix operand order repaired and checked by reading, not compiled, not run on hardware
//
#include <hip/hip_runtime.h>

#define NROWS 131072
#define DDIM  512
#define NBOOK 4
#define KCODES 1024

// 16 codebook rows staged per tile; pad row stride to 1040B (65*16) to spread banks
#define LDS_ROW_BYTES 1040
#define TILE_BYTES    (16 * LDS_ROW_BYTES)
#define NTILES        (KCODES / 16)

typedef __attribute__((ext_vector_type(16))) __bf16 v16bf;
typedef __attribute__((ext_vector_type(8)))  float  v8f;

__device__ __forceinline__ unsigned short f2bf(float f) {
    unsigned u = __float_as_uint(f);
    u += 0x7FFFu + ((u >> 16) & 1u);   // round-to-nearest-even
    return (unsigned short)(u >> 16);
}

// ---------------------------------------------------------------------------
// Pre-pass: fp32 codebooks -> bf16 copy + per-code squared norms.
// One block (128 threads) per code row (NBOOK*KCODES rows of DDIM floats).
// ---------------------------------------------------------------------------
__global__ void rq_prep_codebook(const float* __restrict__ cb,
                                 unsigned short* __restrict__ cb_bf,
                                 float* __restrict__ cnorm) {
    const int row = blockIdx.x;                  // 0 .. NBOOK*KCODES-1
    const int tid = threadIdx.x;                 // 0 .. 127
    const float* src = cb + (size_t)row * DDIM;
    unsigned short* dst = cb_bf + (size_t)row * DDIM;

    float s = 0.f;
    for (int i = tid; i < DDIM; i += 128) {
        float v = src[i];
        s += v * v;
        dst[i] = f2bf(v);
    }
#pragma unroll
    for (int off = 16; off >= 1; off >>= 1) s += __shfl_xor(s, off, 32);

    __shared__ float red[4];
    if ((tid & 31) == 0) red[tid >> 5] = s;
    __syncthreads();
    if (tid == 0) cnorm[row] = red[0] + red[1] + red[2] + red[3];
}

// ---------------------------------------------------------------------------
// One residual-quantization book.
// Grid: NROWS/128 blocks of 256 threads (8 waves). Wave w owns rows
// [block*128 + w*16, +16). A-panel (16 rows x 512 K, bf16) held in VGPRs.
// Codebook swept in 64 tiles of 16 codes; each tile DMA'd to LDS with
// GLOBAL_LOAD_ASYNC_TO_LDS_B128 (ASYNCcnt), double-buffered so the DMA of
// tile nt+1 overlaps the 16 WMMAs of tile nt.
// ---------------------------------------------------------------------------
__global__ void __launch_bounds__(256)
rq_book_kernel(const float* __restrict__ res_in,
               float* __restrict__ res_out,
               const float* __restrict__ cb_f32,
               const unsigned short* __restrict__ cb_bf,
               const float* __restrict__ cnorm,
               float* __restrict__ out_idx,
               int book) {
    __shared__ __align__(16) unsigned char lds[2 * TILE_BYTES];

    const int tid  = threadIdx.x;
    const int wave = tid >> 5;
    const int lane = tid & 31;
    const int half = lane >> 4;    // 0: lanes 0-15, 1: lanes 16-31
    const int lrow = lane & 15;

    const unsigned short* cbb = cb_bf + (size_t)book * KCODES * DDIM;
    const float* cnb = cnorm + (size_t)book * KCODES;
    const float* cb32 = cb_f32 + (size_t)book * KCODES * DDIM;

    const long wave_base = (long)blockIdx.x * 128 + wave * 16;

    // Flat->LDS mapping uses addr[31:0]: low 32 bits of the generic pointer
    // to the __shared__ array give the 0-based LDS byte offset.
    const unsigned lds_base = (unsigned)(uintptr_t)(void*)lds;

    // Per-thread staging slice: 4 x 16B chunks of the 16KB tile.
    const int st_row0 = tid >> 6;          // chunk (tid + i*256): row = c>>6
    const int st_col  = tid & 63;          // 16B chunk within the 1KB row

    // ---- Build A panel: this lane supplies row (wave_base + lrow).
    // ISA 16-bit A layout: element e of kstep t  <->  K = t*32 + 8*half + (e<8 ? e : e+8)
    const float* rsrc = res_in + (wave_base + lrow) * (long)DDIM;
    v16bf a[16];
#pragma unroll
    for (int t = 0; t < 16; t++) {
        const float* p0 = rsrc + t * 32 + half * 8;   // K = t*32+8h .. +7
        const float* p1 = p0 + 16;                    // K = t*32+16+8h .. +7
        float4 f0 = *(const float4*)(p0);
        float4 f1 = *(const float4*)(p0 + 4);
        float4 f2 = *(const float4*)(p1);
        float4 f3 = *(const float4*)(p1 + 4);
        union { v16bf v; unsigned short s[16]; } u;
        u.s[0]  = f2bf(f0.x); u.s[1]  = f2bf(f0.y); u.s[2]  = f2bf(f0.z); u.s[3]  = f2bf(f0.w);
        u.s[4]  = f2bf(f1.x); u.s[5]  = f2bf(f1.y); u.s[6]  = f2bf(f1.z); u.s[7]  = f2bf(f1.w);
        u.s[8]  = f2bf(f2.x); u.s[9]  = f2bf(f2.y); u.s[10] = f2bf(f2.z); u.s[11] = f2bf(f2.w);
        u.s[12] = f2bf(f3.x); u.s[13] = f2bf(f3.y); u.s[14] = f2bf(f3.z); u.s[15] = f2bf(f3.w);
        a[t] = u.v;
    }

    float minv[8];
    int   mini[8];
#pragma unroll
    for (int v = 0; v < 8; v++) { minv[v] = __builtin_inff(); mini[v] = 0; }

    // ---- Async-DMA the first tile into buffer 0.
    {
        const char* gsrc = (const char*)cbb;             // tile 0: codes 0..15
#pragma unroll
        for (int i = 0; i < 4; i++) {
            const int row = st_row0 + i * 4;             // (tid + i*256) >> 6
            unsigned la = lds_base + (unsigned)(row * LDS_ROW_BYTES + st_col * 16);
            unsigned long long ga = (unsigned long long)(uintptr_t)
                (gsrc + (size_t)row * DDIM * 2 + (size_t)st_col * 16);
            asm volatile("global_load_async_to_lds_b128 %0, %1, off"
                         :: "v"(la), "v"(ga) : "memory");
        }
    }

    // ---- Sweep all 1024 codes in tiles of 16 (double-buffered async staging).
    for (int nt = 0; nt < NTILES; nt++) {
        const int n0 = nt * 16;

        // Our batch for tile nt has landed when ASYNCcnt==0.
        asm volatile("s_wait_asynccnt 0x0" ::: "memory");
        __syncthreads();   // all waves' DMA visible; all readers of next buffer done

        // Kick off the DMA for tile nt+1 into the other buffer.
        if (nt + 1 < NTILES) {
            const char* gsrc = (const char*)cbb + (size_t)(n0 + 16) * DDIM * 2;
            const unsigned buf = (unsigned)((nt + 1) & 1) * TILE_BYTES;
#pragma unroll
            for (int i = 0; i < 4; i++) {
                const int row = st_row0 + i * 4;
                unsigned la = lds_base + buf + (unsigned)(row * LDS_ROW_BYTES + st_col * 16);
                unsigned long long ga = (unsigned long long)(uintptr_t)
                    (gsrc + (size_t)row * DDIM * 2 + (size_t)st_col * 16);
                asm volatile("global_load_async_to_lds_b128 %0, %1, off"
                             :: "v"(la), "v"(ga) : "memory");
            }
        }

        // dot-product tile: rows (wave_base..+15) x codes (n0..+15) over D=512
        const unsigned char* buf = lds + (nt & 1) * TILE_BYTES;
        v8f acc = {};
#pragma unroll
        for (int t = 0; t < 16; t++) {
            // B layout: element e <-> K = t*32 + 8*half + (e<8 ? e : e+8); column N = lrow
            const unsigned char* bp = buf + lrow * LDS_ROW_BYTES + t * 64 + half * 16;
            union { v16bf v; uint4 q[2]; } ub;
            ub.q[0] = *(const uint4*)(bp);
            ub.q[1] = *(const uint4*)(bp + 32);
            acc = __builtin_amdgcn_wmma_f32_16x16x32_bf16(
                false, a[t], false, ub.v, (short)0, acc, false, false);
        }

        // d2 (up to the per-row constant) = |C|^2 - 2*dot ; track running argmin.
        const float cn = cnb[n0 + lrow];
        const int   code = n0 + lrow;
#pragma unroll
        for (int v = 0; v < 8; v++) {
            float d2 = __builtin_fmaf(-2.0f, acc[v], cn);
            bool lt = d2 < minv[v];
            mini[v] = lt ? code : mini[v];
            minv[v] = lt ? d2 : minv[v];
        }
    }

    // ---- Cross-lane argmin within each 16-lane half (C/D layout halves).
#pragma unroll
    for (int v = 0; v < 8; v++) {
        float mv = minv[v]; int mi = mini[v];
#pragma unroll
        for (int off = 8; off >= 1; off >>= 1) {
            float ov = __shfl_xor(mv, off, 16);
            int   oi = __shfl_xor(mi, off, 16);
            if (ov < mv || (ov == mv && oi < mi)) { mv = ov; mi = oi; }
        }
        minv[v] = mv; mini[v] = mi;
    }

    // Slot v holds: lanes 0-15 -> row wave_base+v ; lanes 16-31 -> row wave_base+8+v.
    if (lrow == 0) {
        const long rbase = wave_base + half * 8;
#pragma unroll
        for (int v = 0; v < 8; v++)
            out_idx[(rbase + v) * NBOOK + book] = (float)mini[v];
    }

    // ---- Residual update: res_out[row] = res_in[row] - C[idx[row]] (fp32).
#pragma unroll
    for (int r = 0; r < 16; r++) {
        const int src_lane = (r >> 3) * 16;
        const int idx = __shfl(mini[r & 7], src_lane, 32);
        const long row = wave_base + r;
        const float* cptr = cb32 + (size_t)idx * DDIM;
        const float* rin  = res_in  + row * (long)DDIM;
        float*       rout = res_out + row * (long)DDIM;
#pragma unroll
        for (int j = 0; j < 4; j++) {
            const int col = (lane + j * 32) * 4;     // 128 float4 per row / 32 lanes
            float4 rv = *(const float4*)(rin + col);
            float4 cv = *(const float4*)(cptr + col);
            float4 o;
            o.x = rv.x - cv.x; o.y = rv.y - cv.y;
            o.z = rv.z - cv.z; o.w = rv.w - cv.w;
            *(float4*)(rout + col) = o;
        }
    }
}

// ---------------------------------------------------------------------------
extern "C" void kernel_launch(void* const* d_in, const int* in_sizes, int n_in,
                              void* d_out, int out_size, void* d_ws, size_t ws_size,
                              hipStream_t stream) {
    const float* x  = (const float*)d_in[0];   // [N, D] fp32
    const float* cb = (const float*)d_in[1];   // [NBOOK, K, D] fp32

    float* out = (float*)d_out;
    float* out_idx = out;                             // N*NBOOK "float" indices
    float* res     = out + (size_t)NROWS * NBOOK;     // N*D residual (also scratch)

    unsigned short* cb_bf = (unsigned short*)d_ws;    // 4 MB bf16 codebooks
    float* cnorm = (float*)((char*)d_ws + (size_t)NBOOK * KCODES * DDIM * sizeof(unsigned short));

    rq_prep_codebook<<<dim3(NBOOK * KCODES), dim3(128), 0, stream>>>(cb, cb_bf, cnorm);

    for (int b = 0; b < NBOOK; b++) {
        const float* rin = (b == 0) ? x : res;
        rq_book_kernel<<<dim3(NROWS / 128), dim3(256), 0, stream>>>(
            rin, res, cb, cb_bf, cnorm, out_idx, b);
    }
}
